// Document_49323404427377
// MI455X (gfx1250) — matrix operand order repaired and checked by reading, not compiled, not run on hardware
//
#include <hip/hip_runtime.h>

typedef __attribute__((ext_vector_type(16))) __bf16 v16bf;
typedef __attribute__((ext_vector_type(8)))  float  v8f;

#define NS_   100000
#define ND_   10000
#define HH    128
#define ESS_  1600000
#define EDS_  200000
#define KTOT  384   // 128 (agg_ss) + 128 (agg_ds) + 128 (self-loop)
#define KCH   64    // K-chunk staged in LDS per step (2 wmma K-steps)
#define LPAD  72    // padded LDS row stride (bf16 elems): n*36 mod 64 is conflict-free

struct B32x2 { uint4 lo, hi; };   // 32 bytes, bit-castable to v16bf

__device__ inline unsigned short f2bf(float f) {
  unsigned int u = __builtin_bit_cast(unsigned int, f);
  u += 0x7FFFu + ((u >> 16) & 1u);       // round-to-nearest-even
  return (unsigned short)(u >> 16);
}

// ---------------- utility kernels ----------------

__global__ void fill_zero_f32(float* __restrict__ p, size_t n) {
  size_t i = (size_t)blockIdx.x * blockDim.x + threadIdx.x;
  size_t stride = (size_t)gridDim.x * blockDim.x;
  for (; i < n; i += stride) p[i] = 0.0f;
}

__global__ void deg_count(const int* __restrict__ dst, int n, float* __restrict__ deg) {
  int i = blockIdx.x * blockDim.x + threadIdx.x;
  if (i < n) atomicAdd(&deg[dst[i]], 1.0f);
}

__global__ void invert_deg(float* __restrict__ deg, int n) {
  int i = blockIdx.x * blockDim.x + threadIdx.x;
  if (i < n) deg[i] = 1.0f / fmaxf(deg[i], 1.0f);
}

// one wave per edge; each lane moves one float4 (32 lanes * 4 = 128 floats)
__global__ void scatter_add(const float* __restrict__ feat,
                            const int* __restrict__ src, const int* __restrict__ dst,
                            int nedges, float* __restrict__ agg) {
  int e    = (int)(((size_t)blockIdx.x * blockDim.x + threadIdx.x) >> 5);
  int lane = threadIdx.x & 31;
  if (e >= nedges) return;
  int s = src[e], d = dst[e];
  float4 v = ((const float4*)(feat + (size_t)s * HH))[lane];
  float* o = agg + (size_t)d * HH + lane * 4;
  atomicAdd(o + 0, v.x);
  atomicAdd(o + 1, v.y);
  atomicAdd(o + 2, v.z);
  atomicAdd(o + 3, v.w);
}

// X[:, ofs:ofs+128] = bf16(agg * rdeg)   (n = NS*128)
__global__ void build_x_scaled(const float* __restrict__ agg, const float* __restrict__ rdeg,
                               unsigned short* __restrict__ X, int ofs, int n) {
  int idx = blockIdx.x * blockDim.x + threadIdx.x;
  if (idx >= n) return;
  int i = idx >> 7, c = idx & 127;
  X[(size_t)i * KTOT + ofs + c] = f2bf(agg[idx] * rdeg[i]);
}

// X[:,0:128] = bf16(agg_ss * rdeg), X[:,256:384] = bf16(hs)
__global__ void build_x_layer(const float* __restrict__ agg, const float* __restrict__ rdeg,
                              const float* __restrict__ hs,
                              unsigned short* __restrict__ X, int n) {
  int idx = blockIdx.x * blockDim.x + threadIdx.x;
  if (idx >= n) return;
  int i = idx >> 7, c = idx & 127;
  size_t base = (size_t)i * KTOT;
  X[base + c]       = f2bf(agg[idx] * rdeg[i]);
  X[base + 256 + c] = f2bf(hs[idx]);
}

// Wcat[l][n][k] (bf16, [N][K] = pre-transposed): k<128 -> W_rel[l,0][k][n],
// k<256 -> W_rel[l,1][k-128][n], else W_loop[l][k-256][n]
__global__ void prep_w(const float* __restrict__ Wrel, const float* __restrict__ Wloop,
                       unsigned short* __restrict__ Wcat) {
  int idx = blockIdx.x * blockDim.x + threadIdx.x;
  if (idx >= 3 * HH * KTOT) return;
  int l = idx / (HH * KTOT);
  int r = idx % (HH * KTOT);
  int n = r / KTOT;
  int k = r % KTOT;
  float w;
  if (k < 128)      w = Wrel[(((size_t)l * 2 + 0) * HH + k)       * HH + n];
  else if (k < 256) w = Wrel[(((size_t)l * 2 + 1) * HH + (k-128)) * HH + n];
  else              w = Wloop[((size_t)l * HH + (k-256))          * HH + n];
  Wcat[idx] = f2bf(w);
}

// ---------------- WMMA GEMM ----------------
// A fragment, bf16 16x32 (ISA layout): lanes 0-15 row M=lane, K chunks [0..7],[16..23];
// lanes 16-31 row M=lane-16, K chunks [8..15],[24..31].
__device__ inline v16bf load_frag_a(const unsigned short* __restrict__ X, int row,
                                    int kbase, int lane) {
  const unsigned short* p = X + (size_t)row * KTOT + kbase + ((lane < 16) ? 0 : 8);
  B32x2 t;
  t.lo = *(const uint4*)(p);
  t.hi = *(const uint4*)(p + 16);
  return __builtin_bit_cast(v16bf, t);
}

// B fragment from LDS, bf16 32x16: lane holds column nbase+(lane&15);
// lanes 0-15 K chunks [0..7],[16..23]; lanes 16-31 K chunks [8..15],[24..31] (padded [N][KCH] LDS tile)
__device__ inline v16bf lds_frag_b(const unsigned short* sW, int nbase, int klocal, int lane) {
  const unsigned short* p = sW + (nbase + (lane & 15)) * LPAD + klocal
                               + ((lane < 16) ? 0 : 8);
  B32x2 t;
  t.lo = *(const uint4*)(p);
  t.hi = *(const uint4*)(p + 16);
  return __builtin_bit_cast(v16bf, t);
}

// gfx1250 async global->LDS copy, 16B per lane, tracked on ASYNCcnt
__device__ inline void async_copy_b128(unsigned lds_off, const void* gaddr) {
  asm volatile("global_load_async_to_lds_b128 %0, %1, off"
               :: "v"(lds_off), "v"(gaddr)
               : "memory");
}

// stage one [128 x KCH] bf16 weight chunk (padded rows) into LDS: 1024 x 16B,
// 256 threads -> 4 async b128 each
__device__ inline void stage_w_async(unsigned short* dstLds,
                                     const unsigned short* __restrict__ WtChunk,
                                     int tid) {
#pragma unroll
  for (int i = 0; i < 4; ++i) {
    int j  = tid + i * 256;      // uint4 index 0..1023
    int n  = j >> 3;             // weight row (output column)
    int k8 = (j & 7) * 8;        // bf16 offset within chunk row
    unsigned lds_off = (unsigned)(uintptr_t)(dstLds + n * LPAD + k8);
    async_copy_b128(lds_off, WtChunk + (size_t)n * KTOT + k8);
  }
}

// 256 threads = 8 waves; block computes 128 rows x 128 cols; wave = 16 rows x 128 cols
__global__ __launch_bounds__(256) void gemm_bias_relu(
    const unsigned short* __restrict__ X, const unsigned short* __restrict__ Wt,
    const float* __restrict__ bias, float* __restrict__ out, int nrows) {
  __shared__ unsigned short sW[2][HH * LPAD];   // 2 x 18 KB double buffer

  int tid  = threadIdx.x;
  int wave = tid >> 5;
  int lane = tid & 31;
  int row0 = blockIdx.x * 128 + wave * 16;

  v8f acc[8];
#pragma unroll
  for (int i = 0; i < 8; ++i) acc[i] = v8f{0.f,0.f,0.f,0.f,0.f,0.f,0.f,0.f};

  int arow = row0 + (lane & 15);
  if (arow >= nrows) arow = nrows - 1;          // clamp: EXEC must stay all-ones for WMMA
  const unsigned short* aptr = X + (size_t)arow * KTOT;

  stage_w_async(&sW[0][0], Wt, tid);            // chunk 0 -> buffer 0

#pragma unroll
  for (int c = 0; c < KTOT / KCH; ++c) {        // 6 chunks
    const int buf = c & 1;
    // issue next chunk into the other buffer (protected by end-of-iter barrier below)
    if (c + 1 < KTOT / KCH)
      stage_w_async(&sW[buf ^ 1][0], Wt + (c + 1) * KCH, tid);

    // overlap: A fragments for this chunk from global while async copy lands
    v16bf a0 = load_frag_a(X, arow, c * KCH,      lane);
    v16bf a1 = load_frag_a(X, arow, c * KCH + 32, lane);
    if (c + 2 < KTOT / KCH)                      // gfx1250 global_prefetch_b8
      __builtin_prefetch(aptr + (c + 2) * KCH, 0, 3);

    // wait for chunk c's 4 async ops (in-order; next stage's 4 may remain in flight)
    if (c + 1 < KTOT / KCH)
      asm volatile("s_wait_asynccnt 0x4" ::: "memory");
    else
      asm volatile("s_wait_asynccnt 0x0" ::: "memory");
    __syncthreads();                             // chunk c visible from all waves

    const unsigned short* sWb = &sW[buf][0];
    int kl0 = 0;
#pragma unroll
    for (int nt = 0; nt < 8; ++nt) {
      v16bf b0 = lds_frag_b(sWb, nt * 16, kl0,      lane);
      v16bf b1 = lds_frag_b(sWb, nt * 16, kl0 + 32, lane);
      acc[nt] = __builtin_amdgcn_wmma_f32_16x16x32_bf16(
          false, a0, false, b0, (short)0, acc[nt], false, false);
      acc[nt] = __builtin_amdgcn_wmma_f32_16x16x32_bf16(
          false, a1, false, b1, (short)0, acc[nt], false, false);
    }
    if (c + 1 < KTOT / KCH)
      __syncthreads();                           // all waves done reading buf before reuse
  }

  // C/D layout: VGPR i -> M = i (lanes 0-15) or 8+i (lanes 16-31); N = lane&15
  int mofs = (lane < 16) ? 0 : 8;
  int col0 = lane & 15;
#pragma unroll
  for (int nt = 0; nt < 8; ++nt) {
    int col = nt * 16 + col0;
    float bv = bias[col];
#pragma unroll
    for (int i = 0; i < 8; ++i) {
      int row = row0 + mofs + i;
      if (row < nrows) {
        float v = acc[nt][i] + bv;
        out[(size_t)row * HH + col] = v > 0.f ? v : 0.f;
      }
    }
  }
}

// ---------------- driver ----------------

extern "C" void kernel_launch(void* const* d_in, const int* in_sizes, int n_in,
                              void* d_out, int out_size, void* d_ws, size_t ws_size,
                              hipStream_t stream) {
  const float* s_feat   = (const float*)d_in[0];
  const float* doc_feat = (const float*)d_in[1];
  const float* W_rel    = (const float*)d_in[2];
  const float* W_loop   = (const float*)d_in[3];
  const float* bias     = (const float*)d_in[4];
  const int*   ss_src   = (const int*)d_in[5];
  const int*   ss_dst   = (const int*)d_in[6];
  const int*   ds_src   = (const int*)d_in[7];
  const int*   ds_dst   = (const int*)d_in[8];
  float* out = (float*)d_out;

  size_t off = 0;
  char* ws = (char*)d_ws;
  auto take = [&](size_t bytes) -> void* {
    void* p = ws + off;
    off += (bytes + 255) & ~(size_t)255;
    return p;
  };
  float*          rdeg_ss = (float*)take((size_t)NS_ * 4);
  float*          rdeg_ds = (float*)take((size_t)NS_ * 4);
  float*          agg     = (float*)take((size_t)NS_ * HH * 4);
  unsigned short* X       = (unsigned short*)take((size_t)NS_ * KTOT * 2);
  float*          hsA     = (float*)take((size_t)NS_ * HH * 4);
  float*          hsB     = (float*)take((size_t)NS_ * HH * 4);
  unsigned short* Wcat    = (unsigned short*)take((size_t)3 * HH * KTOT * 2);

  const int NF = NS_ * HH;                 // 12.8M feature elements
  dim3 b256(256);

  // degrees -> reciprocal right-norm factors (constant across layers)
  fill_zero_f32<<<2048, b256, 0, stream>>>(rdeg_ss, (size_t)NS_);
  fill_zero_f32<<<2048, b256, 0, stream>>>(rdeg_ds, (size_t)NS_);
  deg_count<<<(ESS_ + 255) / 256, b256, 0, stream>>>(ss_dst, ESS_, rdeg_ss);
  deg_count<<<(EDS_ + 255) / 256, b256, 0, stream>>>(ds_dst, EDS_, rdeg_ds);
  invert_deg<<<(NS_ + 255) / 256, b256, 0, stream>>>(rdeg_ss, NS_);
  invert_deg<<<(NS_ + 255) / 256, b256, 0, stream>>>(rdeg_ds, NS_);

  // doc->sentence aggregation: layer-invariant, compute once into X[:,128:256]
  fill_zero_f32<<<4096, b256, 0, stream>>>(agg, (size_t)NF);
  scatter_add<<<(EDS_ * 32 + 255) / 256, b256, 0, stream>>>(doc_feat, ds_src, ds_dst, EDS_, agg);
  build_x_scaled<<<(NF + 255) / 256, b256, 0, stream>>>(agg, rdeg_ds, X, 128, NF);

  // weights: bf16, [N][K] pre-transposed, K = [W_rel0 | W_rel1 | W_loop]
  prep_w<<<(3 * HH * KTOT + 255) / 256, b256, 0, stream>>>(W_rel, W_loop, Wcat);

  const int gemm_blocks = (NS_ + 127) / 128;
  const float* hs_cur = s_feat;
  for (int l = 0; l < 3; ++l) {
    fill_zero_f32<<<4096, b256, 0, stream>>>(agg, (size_t)NF);
    scatter_add<<<(ESS_ * 32 + 255) / 256, b256, 0, stream>>>(hs_cur, ss_src, ss_dst, ESS_, agg);
    build_x_layer<<<(NF + 255) / 256, b256, 0, stream>>>(agg, rdeg_ss, hs_cur, X, NF);
    float* hs_next = (l == 2) ? out : ((l == 0) ? hsA : hsB);
    gemm_bias_relu<<<gemm_blocks, b256, 0, stream>>>(
        X, Wcat + (size_t)l * HH * KTOT, bias + (size_t)l * HH, hs_next, NS_);
    hs_cur = hs_next;
  }
}
